// MultiHeadAttention_8564164789043
// MI455X (gfx1250) — matrix operand order repaired and checked by reading, not compiled, not run on hardware
//
#include <hip/hip_runtime.h>
#include <stdint.h>

// ---------------------------------------------------------------------------
// MI455X (gfx1250) MHA, bf16 WMMA + fused flash-style attention.
//   1) q/k/v projections  (fp32 -> bf16, WMMA GEMM + bias)
//   2) fused attention: 3 passes over K tiles (max, sum, emit):
//        writes normalized P to d_out exactly once (512MB = the mandatory
//        traffic), accumulates ctx = P*V from LDS.  K tiles staged with
//        double-buffered gfx1250 async global->LDS copies (ASYNCcnt),
//        overlapping DMA of tile t+1 with WMMA on tile t.
//   3) out = ctx*Wo + bo  (bf16 x f32 -> fp32, WMMA GEMM + bias)
// ---------------------------------------------------------------------------

typedef __bf16 bf16_t;
typedef __attribute__((ext_vector_type(16))) __bf16 v16bf;
typedef __attribute__((ext_vector_type(8)))  __bf16 v8bf;
typedef __attribute__((ext_vector_type(4)))  __bf16 v4bf;
typedef __attribute__((ext_vector_type(8)))  float  v8f;

static __device__ __forceinline__ bf16_t f2bf(float f) {
  unsigned u = __builtin_bit_cast(unsigned, f);
  unsigned r = u + 0x7FFFu + ((u >> 16) & 1u);   // round-to-nearest-even
  unsigned short h = (unsigned short)(r >> 16);
  return __builtin_bit_cast(bf16_t, h);
}

// gfx1250 async global->LDS copy (16B per lane), tracked by ASYNCcnt.
static __device__ __forceinline__ void async_copy16(bf16_t* lds, const bf16_t* g) {
  unsigned off = (unsigned)(uintptr_t)lds;       // low 32 bits = LDS offset
  asm volatile("global_load_async_to_lds_b128 %0, %1, off"
               :: "v"(off), "v"(g) : "memory");
}
static __device__ __forceinline__ void wait_async0() {
  asm volatile("s_wait_asynccnt 0x0" ::: "memory");
}
// Async loads complete in order: <=2 outstanding leaves only the 2 ops of the
// most recently issued tile in flight.
static __device__ __forceinline__ void wait_async_le2() {
  asm volatile("s_wait_asynccnt 0x2" ::: "memory");
}

// ---------------------------------------------------------------------------
// Generic LDS-staged bf16 WMMA GEMM:  C = alpha * (A x B) [+ bias]
// ---------------------------------------------------------------------------
template <int BM, int BN, int WM, int WN,
          bool A_F32, bool B_F32, bool B_T, bool OUT_F32, bool BIAS>
__global__ __launch_bounds__(256)
void wmma_gemm(const void* __restrict__ Ap, const void* __restrict__ Bp,
               const float* __restrict__ bias, void* __restrict__ Cp,
               int M, int N, int K, int lda, int ldb, int ldc,
               long aBatch, long bBatch, long cBatch, float alpha)
{
  static_assert((BM / WM) * (BN / WN) == 8, "8 waves per block");
  constexpr int BK   = 32;
  constexpr int PADK = BK + 8;
  constexpr int TM = WM / 16, TN = WN / 16;
  constexpr int WAVES_N = BN / WN;

  __shared__ alignas(16) bf16_t lA[BM * PADK];
  __shared__ alignas(16) bf16_t lB[BN * PADK];

  const int tid  = (int)threadIdx.x;
  const int wave = tid >> 5;
  const int lane = tid & 31;
  const int l16  = lane & 15;
  const int lhi  = lane >> 4;

  const int wm = (wave / WAVES_N) * WM;
  const int wn = (wave % WAVES_N) * WN;
  const int m0 = (int)blockIdx.y * BM;
  const int n0 = (int)blockIdx.x * BN;
  const int b  = (int)blockIdx.z;

  const float*  Af = (const float*) Ap + (long)b * aBatch;
  const bf16_t* Ah = (const bf16_t*)Ap + (long)b * aBatch;
  const float*  Bf = (const float*) Bp + (long)b * bBatch;
  const bf16_t* Bh = (const bf16_t*)Bp + (long)b * bBatch;
  float*  Cf = (float*) Cp + (long)b * cBatch;
  bf16_t* Ch = (bf16_t*)Cp + (long)b * cBatch;

  v8f acc[TM][TN];
  for (int i = 0; i < TM; ++i)
    for (int j = 0; j < TN; ++j)
      for (int e = 0; e < 8; ++e) acc[i][j][e] = 0.0f;

  for (int kt = 0; kt < K; kt += BK) {
    if (kt + BK < K) {
      const int pr = m0 + (tid & (BM - 1));
      if constexpr (A_F32) __builtin_prefetch(&Af[(long)pr * lda + kt + BK], 0, 1);
      else                 __builtin_prefetch(&Ah[(long)pr * lda + kt + BK], 0, 1);
    }
    // ---- stage A tile (vectorized, fp32 converted to bf16) ----
    if constexpr (A_F32) {
      for (int idx = tid; idx < BM * (BK / 4); idx += 256) {
        const int r = idx / (BK / 4), c = (idx % (BK / 4)) * 4;
        const float4 f = *(const float4*)&Af[(long)(m0 + r) * lda + kt + c];
        v4bf t; t[0] = f2bf(f.x); t[1] = f2bf(f.y); t[2] = f2bf(f.z); t[3] = f2bf(f.w);
        *(v4bf*)&lA[r * PADK + c] = t;
      }
    } else {
      for (int idx = tid; idx < BM * (BK / 8); idx += 256) {
        const int r = idx / (BK / 8), c = (idx % (BK / 8)) * 8;
        *(v8bf*)&lA[r * PADK + c] = *(const v8bf*)&Ah[(long)(m0 + r) * lda + kt + c];
      }
    }
    // ---- stage B tile as [n][k] ----
    if constexpr (B_T) {
      if constexpr (B_F32) {
        for (int idx = tid; idx < BN * (BK / 4); idx += 256) {
          const int n = idx / (BK / 4), c = (idx % (BK / 4)) * 4;
          const float4 f = *(const float4*)&Bf[(long)(n0 + n) * ldb + kt + c];
          v4bf t; t[0] = f2bf(f.x); t[1] = f2bf(f.y); t[2] = f2bf(f.z); t[3] = f2bf(f.w);
          *(v4bf*)&lB[n * PADK + c] = t;
        }
      } else {
        for (int idx = tid; idx < BN * (BK / 8); idx += 256) {
          const int n = idx / (BK / 8), c = (idx % (BK / 8)) * 8;
          *(v8bf*)&lB[n * PADK + c] = *(const v8bf*)&Bh[(long)(n0 + n) * ldb + kt + c];
        }
      }
    } else {
      if constexpr (B_F32) {
        for (int idx = tid; idx < BK * (BN / 4); idx += 256) {
          const int k = idx / (BN / 4), n4 = (idx % (BN / 4)) * 4;
          const float4 f = *(const float4*)&Bf[(long)(kt + k) * ldb + n0 + n4];
          const float fv[4] = { f.x, f.y, f.z, f.w };
          for (int e = 0; e < 4; ++e) lB[(n4 + e) * PADK + k] = f2bf(fv[e]);
        }
      } else {
        for (int idx = tid; idx < BK * (BN / 8); idx += 256) {
          const int k = idx / (BN / 8), n8 = (idx % (BN / 8)) * 8;
          const v8bf v = *(const v8bf*)&Bh[(long)(kt + k) * ldb + n0 + n8];
          for (int e = 0; e < 8; ++e) lB[(n8 + e) * PADK + k] = v[e];
        }
      }
    }
    __syncthreads();

    v16bf afr[TM];
    for (int i = 0; i < TM; ++i) {
      const bf16_t* p = &lA[(wm + i * 16 + l16) * PADK + lhi * 8];
      v8bf lo = *(const v8bf*)(p);
      v8bf hi = *(const v8bf*)(p + 16);
      for (int e = 0; e < 8; ++e) { afr[i][e] = lo[e]; afr[i][8 + e] = hi[e]; }
    }
    v16bf bfr[TN];
    for (int j = 0; j < TN; ++j) {
      const bf16_t* p = &lB[(wn + j * 16 + l16) * PADK + lhi * 8];
      v8bf lo = *(const v8bf*)(p);
      v8bf hi = *(const v8bf*)(p + 16);
      for (int e = 0; e < 8; ++e) { bfr[j][e] = lo[e]; bfr[j][8 + e] = hi[e]; }
    }
    for (int i = 0; i < TM; ++i)
      for (int j = 0; j < TN; ++j)
        acc[i][j] = __builtin_amdgcn_wmma_f32_16x16x32_bf16(
            false, afr[i], false, bfr[j], (short)0, acc[i][j], false, false);
    __syncthreads();
  }

  for (int j = 0; j < TN; ++j) {
    const int col = n0 + wn + j * 16 + l16;
    const float bv = BIAS ? bias[col] : 0.0f;
    for (int i = 0; i < TM; ++i) {
      for (int r = 0; r < 8; ++r) {
        const int row = m0 + wm + i * 16 + r + lhi * 8;
        const float v = acc[i][j][r] * alpha + bv;
        if constexpr (OUT_F32) Cf[(long)row * ldc + col] = v;
        else                   Ch[(long)row * ldc + col] = f2bf(v);
      }
    }
  }
}

// ---------------------------------------------------------------------------
// Fused attention: per block = one (head-batch, 64-query tile).
// 3 passes over 32 K-tiles: rowmax, rowsum, emit P (once) + accumulate P*V.
// K (and V) tiles double-buffered; async DMA of tile t+1 overlaps tile t.
// ---------------------------------------------------------------------------
__global__ __launch_bounds__(256)
void flash_attn(const bf16_t* __restrict__ qb, const bf16_t* __restrict__ kb,
                const bf16_t* __restrict__ vb, float* __restrict__ att,
                bf16_t* __restrict__ ctxb)
{
  constexpr int L = 2048, DH = 64, QT = 64, KT = 64, NT = L / KT;
  constexpr int PAD = DH + 8;        // padded LDS row stride (bf16)
  constexpr float SCALE = 0.125f;    // 1/sqrt(64)

  __shared__ alignas(16) bf16_t Qs[QT * PAD];
  __shared__ alignas(16) bf16_t Ks[2][KT * PAD];
  __shared__ alignas(16) bf16_t Vt[2][DH * PAD];  // Vt[d][key]
  __shared__ alignas(16) bf16_t Ps[QT * PAD];
  __shared__ float rmax2[2][QT];
  __shared__ float rsum2[2][QT];

  const int tid  = (int)threadIdx.x;
  const int wave = tid >> 5;
  const int lane = tid & 31;
  const int l16  = lane & 15;
  const int lhi  = lane >> 4;
  const int wm   = (wave >> 1) * 16;   // S/ctx tile row base (4 waves in M)
  const int wn   = (wave & 1) * 32;    // S/ctx tile col base (2 waves in N)
  const int bh   = (int)blockIdx.y;
  const int q0   = (int)blockIdx.x * QT;

  const bf16_t* Qg = qb + ((long)bh * L + q0) * DH;
  const bf16_t* Kg = kb + (long)bh * L * DH;
  const bf16_t* Vg = vb + (long)bh * L * DH;
  float*        Pg = att + (long)bh * L * L;

  // per-thread chunk coordinates (64x64 tile = 512 16B chunks, 2 per thread)
  const int cr = tid >> 3;             // rows 0..31 (and +32)
  const int cc = (tid & 7) * 8;        // col 0..56 step 8

  // Q tile: async global->LDS (stays resident all kernel)
  async_copy16(&Qs[cr * PAD + cc],        &Qg[(long)cr * DH + cc]);
  async_copy16(&Qs[(cr + 32) * PAD + cc], &Qg[(long)(cr + 32) * DH + cc]);

  auto issueK = [&](int t, int buf) {
    const bf16_t* base = &Kg[(long)(t * KT) * DH];
    async_copy16(&Ks[buf][cr * PAD + cc],        &base[(long)cr * DH + cc]);
    async_copy16(&Ks[buf][(cr + 32) * PAD + cc], &base[(long)(cr + 32) * DH + cc]);
  };
  auto loadV = [&](int t, int buf) {            // transposed: Vt[d][key]
    const bf16_t* base = &Vg[(long)(t * KT) * DH];
    const v8bf v0 = *(const v8bf*)&base[(long)cr * DH + cc];
    const v8bf v1 = *(const v8bf*)&base[(long)(cr + 32) * DH + cc];
#pragma unroll
    for (int e = 0; e < 8; ++e) {
      Vt[buf][(cc + e) * PAD + cr]      = v0[e];
      Vt[buf][(cc + e) * PAD + cr + 32] = v1[e];
    }
  };
  // S fragments for this wave's 16x32 slice of the 64x64 S tile.
  auto computeS = [&](v8f* a2, const bf16_t* kbuf) {
    for (int j = 0; j < 2; ++j)
      for (int e = 0; e < 8; ++e) a2[j][e] = 0.0f;
    for (int ks = 0; ks < DH; ks += 32) {
      const bf16_t* ap = &Qs[(wm + l16) * PAD + ks + lhi * 8];
      v8bf alo = *(const v8bf*)(ap), ahi = *(const v8bf*)(ap + 16);
      v16bf a;
      for (int e = 0; e < 8; ++e) { a[e] = alo[e]; a[8 + e] = ahi[e]; }
      for (int j = 0; j < 2; ++j) {
        const bf16_t* bp = &kbuf[(wn + j * 16 + l16) * PAD + ks + lhi * 8];
        v8bf blo = *(const v8bf*)(bp), bhi = *(const v8bf*)(bp + 16);
        v16bf bb;
        for (int e = 0; e < 8; ++e) { bb[e] = blo[e]; bb[8 + e] = bhi[e]; }
        a2[j] = __builtin_amdgcn_wmma_f32_16x16x32_bf16(
            false, a, false, bb, (short)0, a2[j], false, false);
      }
    }
  };

  // ---------- pass 1: row max ----------
  issueK(0, 0);
  float rmax[8];
#pragma unroll
  for (int e = 0; e < 8; ++e) rmax[e] = -3.402823466e38f;
  for (int t = 0; t < NT; ++t) {
    const int cur = t & 1;
    if (t + 1 < NT) { issueK(t + 1, cur ^ 1); wait_async_le2(); }
    else            { wait_async0(); }
    __syncthreads();                       // tile t visible to all waves
    v8f s[2]; computeS(s, Ks[cur]);
#pragma unroll
    for (int j = 0; j < 2; ++j)
#pragma unroll
      for (int e = 0; e < 8; ++e) rmax[e] = fmaxf(rmax[e], s[j][e] * SCALE);
    __syncthreads();                       // reads done before buffer reuse
  }
#pragma unroll
  for (int e = 0; e < 8; ++e)
    for (int m = 1; m < 16; m <<= 1) rmax[e] = fmaxf(rmax[e], __shfl_xor(rmax[e], m, 32));
  if (l16 == 0)
#pragma unroll
    for (int e = 0; e < 8; ++e) rmax2[wave & 1][wm + lhi * 8 + e] = rmax[e];
  __syncthreads();
  float mrow[8];
#pragma unroll
  for (int e = 0; e < 8; ++e)
    mrow[e] = fmaxf(rmax2[0][wm + lhi * 8 + e], rmax2[1][wm + lhi * 8 + e]);

  // ---------- pass 2: row sum ----------
  issueK(0, 0);
  float rsum[8];
#pragma unroll
  for (int e = 0; e < 8; ++e) rsum[e] = 0.0f;
  for (int t = 0; t < NT; ++t) {
    const int cur = t & 1;
    if (t + 1 < NT) { issueK(t + 1, cur ^ 1); wait_async_le2(); }
    else            { wait_async0(); }
    __syncthreads();
    v8f s[2]; computeS(s, Ks[cur]);
#pragma unroll
    for (int j = 0; j < 2; ++j)
#pragma unroll
      for (int e = 0; e < 8; ++e) rsum[e] += __expf(s[j][e] * SCALE - mrow[e]);
    __syncthreads();
  }
#pragma unroll
  for (int e = 0; e < 8; ++e)
    for (int m = 1; m < 16; m <<= 1) rsum[e] += __shfl_xor(rsum[e], m, 32);
  if (l16 == 0)
#pragma unroll
    for (int e = 0; e < 8; ++e) rsum2[wave & 1][wm + lhi * 8 + e] = rsum[e];
  __syncthreads();
  float rinv[8];
#pragma unroll
  for (int e = 0; e < 8; ++e)
    rinv[e] = 1.0f / (rsum2[0][wm + lhi * 8 + e] + rsum2[1][wm + lhi * 8 + e]);

  // ---------- pass 3: emit normalized P + accumulate ctx = P*V ----------
  issueK(0, 0);
  v8f cacc[2];
  for (int j = 0; j < 2; ++j)
    for (int e = 0; e < 8; ++e) cacc[j][e] = 0.0f;
  for (int t = 0; t < NT; ++t) {
    const int cur = t & 1;
    if (t + 1 < NT) issueK(t + 1, cur ^ 1);
    loadV(t, cur);
    if (t + 1 < NT) wait_async_le2(); else wait_async0();
    __syncthreads();                       // Ks(t), Vt(t) visible
    v8f s[2]; computeS(s, Ks[cur]);
#pragma unroll
    for (int j = 0; j < 2; ++j) {
      const int col = wn + j * 16 + l16;
#pragma unroll
      for (int e = 0; e < 8; ++e) {
        const int row = wm + e + lhi * 8;
        const float p = __expf(s[j][e] * SCALE - mrow[e]) * rinv[e];
        Pg[(long)(q0 + row) * L + t * KT + col] = p;   // write P exactly once
        Ps[row * PAD + col] = f2bf(p);
      }
    }
    __syncthreads();                       // Ps visible; Ks reads done
    // PV: A from Ps (this wave's rows), B from Vt
    for (int ks = 0; ks < KT; ks += 32) {
      const bf16_t* ap = &Ps[(wm + l16) * PAD + ks + lhi * 8];
      v8bf alo = *(const v8bf*)(ap), ahi = *(const v8bf*)(ap + 16);
      v16bf a;
      for (int e = 0; e < 8; ++e) { a[e] = alo[e]; a[8 + e] = ahi[e]; }
      for (int j = 0; j < 2; ++j) {
        const bf16_t* bp = &Vt[cur][(wn + j * 16 + l16) * PAD + ks + lhi * 8];
        v8bf blo = *(const v8bf*)(bp), bhi = *(const v8bf*)(bp + 16);
        v16bf bb;
        for (int e = 0; e < 8; ++e) { bb[e] = blo[e]; bb[8 + e] = bhi[e]; }
        cacc[j] = __builtin_amdgcn_wmma_f32_16x16x32_bf16(
            false, a, false, bb, (short)0, cacc[j], false, false);
      }
    }
  }
  // write ctx tile (bf16, raw [32][2048][64] layout)
#pragma unroll
  for (int j = 0; j < 2; ++j) {
    const int col = wn + j * 16 + l16;
#pragma unroll
    for (int e = 0; e < 8; ++e) {
      const int row = q0 + wm + e + lhi * 8;
      ctxb[((long)bh * L + row) * DH + col] = f2bf(cacc[j][e]);
    }
  }
}

// ---------------------------------------------------------------------------
extern "C" void kernel_launch(void* const* d_in, const int* in_sizes, int n_in,
                              void* d_out, int out_size, void* d_ws, size_t ws_size,
                              hipStream_t stream)
{
  (void)in_sizes; (void)n_in; (void)out_size; (void)ws_size;
  constexpr int B = 2, L = 2048, D = 1024, H = 16;
  constexpr int MR = B * L;         // 4096 projection rows
  constexpr int BH = B * H;         // 32 head-batches

  const float* key   = (const float*)d_in[0];
  const float* value = (const float*)d_in[1];
  const float* query = (const float*)d_in[2];
  const float* Wk = (const float*)d_in[3];
  const float* bk = (const float*)d_in[4];
  const float* Wv = (const float*)d_in[5];
  const float* bv = (const float*)d_in[6];
  const float* Wq = (const float*)d_in[7];
  const float* bq = (const float*)d_in[8];
  const float* Wo = (const float*)d_in[9];
  const float* bo = (const float*)d_in[10];

  bf16_t* qb   = (bf16_t*)d_ws;
  bf16_t* kb   = qb + (long)MR * D;
  bf16_t* vb   = kb + (long)MR * D;
  bf16_t* ctxb = vb + (long)MR * D;

  float* outp = (float*)d_out;              // context [B,L,D] fp32
  float* att  = outp + (long)MR * D;        // attention [32,2048,2048] fp32

  const dim3 blk(256);

  // 1) projections
  {
    const dim3 grd(D / 128, MR / 128, 1);
    wmma_gemm<128,128,32,64, true,true,false,false,true><<<grd, blk, 0, stream>>>(
        query, Wq, bq, qb, MR, D, D, D, D, D, 0, 0, 0, 1.0f);
    wmma_gemm<128,128,32,64, true,true,false,false,true><<<grd, blk, 0, stream>>>(
        key,   Wk, bk, kb, MR, D, D, D, D, D, 0, 0, 0, 1.0f);
    wmma_gemm<128,128,32,64, true,true,false,false,true><<<grd, blk, 0, stream>>>(
        value, Wv, bv, vb, MR, D, D, D, D, D, 0, 0, 0, 1.0f);
  }

  // 2) fused attention (scores + softmax + PV), P written once
  flash_attn<<<dim3(L / 64, BH), blk, 0, stream>>>(qb, kb, vb, att, ctxb);

  // 3) output projection
  {
    const dim3 grd(D / 128, MR / 128, 1);
    wmma_gemm<128,128,32,64, false,true,false,true,true><<<grd, blk, 0, stream>>>(
        ctxb, Wo, bo, outp, MR, D, D, D, D, D, 0, 0, 0, 1.0f);
  }
}
